// QuantumFusionLayer_62423054680454
// MI455X (gfx1250) — compile-verified, matrix operand order checked by reading, not run on hardware
//
#include <hip/hip_runtime.h>

typedef _Float16 h16;
typedef __attribute__((ext_vector_type(16))) _Float16 v16h;
typedef __attribute__((ext_vector_type(8)))  _Float16 v8h;
typedef __attribute__((ext_vector_type(8)))  float    v8f;
typedef __attribute__((ext_vector_type(4)))  unsigned int u32x4;
typedef __attribute__((ext_vector_type(8)))  int i32x8;
typedef __attribute__((ext_vector_type(4)))  int i32x4;

// ---------------------------------------------------------------- helpers ---

__device__ __forceinline__ v8f wmma16(v16h a, v16h b, v8f c) {
  // D = A(16x32 f16) * B(32x16 f16) + C(16x16 f32)
  return __builtin_amdgcn_wmma_f32_16x16x32_f16(false, a, false, b, (short)0, c,
                                                false, false);
}

// A-fragment from a row-major f16 matrix in LDS (stride S halfs).
// Caller passes row = mt*16 + (lane&15), kbase = kt*32 + ((lane&16)?8:0).
// Lane layout per ISA 7.12.2 (16-bit A 16x32): lanes<16 hold K {0-7,16-23},
// lanes>=16 hold K {8-15,24-31}.
__device__ __forceinline__ v16h loadA_rm(const h16* __restrict__ base, int S,
                                         int row, int kbase) {
  const h16* p = base + row * S + kbase;
  v8h lo = *(const v8h*)p;        // k + 0..7
  v8h hi = *(const v8h*)(p + 16); // k + 16..23
  v16h r;
#pragma unroll
  for (int i = 0; i < 8; i++) { r[i] = lo[i]; r[i + 8] = hi[i]; }
  return r;
}

// B-fragment from pre-packed weights: contiguous 16 halfs per lane.
__device__ __forceinline__ v16h loadB_prep(const h16* __restrict__ w, int kt,
                                           int nt, int lane) {
  return *(const v16h*)(w + ((((kt << 4) + nt) << 5) + lane) * 16);
}

__device__ __forceinline__ float wsum(float v) {
#pragma unroll
  for (int m = 16; m > 0; m >>= 1) v += __shfl_xor(v, m, 32);
  return v;
}
__device__ __forceinline__ float wmaxr(float v) {
#pragma unroll
  for (int m = 16; m > 0; m >>= 1) v = fmaxf(v, __shfl_xor(v, m, 32));
  return v;
}

// -------------------------------------------------------- weight repacking --

// dst linear index t = ((kt*16 + nt)*32 + lane)*16 + e
// lane<16: e<8 -> k=e,  e>=8 -> k=e+8 ; lane>=16: +8.  col n = nt*16 + (lane&15)
__global__ void prep_w(const float* __restrict__ src, h16* __restrict__ dst,
                       int K) {
  int t = blockIdx.x * 256 + threadIdx.x;
  if (t >= K * 256) return;
  int e = t & 15, lane = (t >> 4) & 31, nt = (t >> 9) & 15, kt = t >> 13;
  int n = nt * 16 + (lane & 15);
  int k = kt * 32 + ((e < 8) ? e : (e + 8)) + ((lane & 16) ? 8 : 0);
  dst[t] = (h16)src[k * 256 + n];
}

// ------------------------------------------------------------- encoders -----
// 16 rows per workgroup, 512 threads = 16 waves (wave w owns output ntile w).
// out = gelu(LN(x @ W + b)) + pos  -> X (f16, row-major, 256 cols)

__global__ void __launch_bounds__(512)
encoder_kernel(const float* __restrict__ in, int Din, int tokBase, int Nmod,
               const h16* __restrict__ Wp, const float* __restrict__ bias,
               const float* __restrict__ g, const float* __restrict__ beta,
               const float* __restrict__ pos, h16* __restrict__ X) {
  __shared__ h16  As[16 * 512];
  __shared__ float Cl[16 * 256];
  int tid = threadIdx.x, lane = tid & 31, wave = tid >> 5;
  int lrow = lane & 15, hi8 = (lane & 16) ? 8 : 0;
  long long rowBase = (long long)blockIdx.x * 16;

  const float* src = in + rowBase * Din;
  for (int i = tid; i < 16 * Din; i += 512) As[i] = (h16)src[i];
  __syncthreads();

  int nt = wave;
  int col = nt * 16 + lrow;
  float bv = bias[col];
  v8f acc;
#pragma unroll
  for (int r = 0; r < 8; r++) acc[r] = bv;
  int ksteps = Din >> 5;
  for (int kt = 0; kt < ksteps; ++kt) {
    v16h a = loadA_rm(As, Din, lrow, kt * 32 + hi8);
    v16h b = loadB_prep(Wp, kt, nt, lane);
    acc = wmma16(a, b, acc);
  }
#pragma unroll
  for (int r = 0; r < 8; r++) Cl[(hi8 + r) * 256 + col] = acc[r];
  __syncthreads();

  // LayerNorm + exact GELU + positional, wave w handles row w
  int r = wave;
  long long rowG = rowBase + r;
  int bidx = (int)(rowG / Nmod);
  int tok = (int)(rowG % Nmod);
  int c0 = lane * 8;
  float v[8], s = 0.f;
#pragma unroll
  for (int i = 0; i < 8; i++) { v[i] = Cl[r * 256 + c0 + i]; s += v[i]; }
  float mean = wsum(s) * (1.f / 256.f);
  float vs = 0.f;
#pragma unroll
  for (int i = 0; i < 8; i++) { float d = v[i] - mean; vs += d * d; }
  float rstd = rsqrtf(wsum(vs) * (1.f / 256.f) + 1e-5f);
  h16* dst = X + ((long long)bidx * 100 + tokBase + tok) * 256;
  const float* pp = pos + (tokBase + tok) * 256;
#pragma unroll
  for (int i = 0; i < 8; i++) {
    int c = c0 + i;
    float y = (v[i] - mean) * rstd * g[c] + beta[c];
    float ge = 0.5f * y * (1.f + erff(y * 0.70710678118f));
    dst[c] = (h16)(ge + pp[c]);
  }
}

// ---------------------------------------------------- fused attention block --
// One workgroup per batch element. N=100 padded to 112 (7 M-tiles).
// LDS byte map (dynamic):
//   Qb  @      0  112x256 f16 (Q, later OUT)
//   Kb  @  57344  112x256 f16 (K, later x1, later FUSED)
//   Vt  @ 114688  256x128 f16 (V transposed, later x2 row-major)
//   Xl  @ 180224  112x256 f16 (X, later scores f32 112x112)
//   Pb  @ 237568  112x128 f16 (softmax probs)
//   AO  @ 266240  112x256 f16 (attention output)
//   red @ 323584  16 f32
#define FUSED_SMEM 323840

__global__ void __launch_bounds__(512)
fused_kernel(const h16* __restrict__ X, const h16* Wq, const h16* Wk,
             const h16* Wv, const h16* Wo, const h16* Wl, const h16* Wg,
             const h16* Wz, const float* qb, const float* kb, const float* vb,
             const float* ob, const float* strengthP, const float* lng,
             const float* lnb, const float* zb, float* __restrict__ out,
             float* __restrict__ entPart) {
  extern __shared__ char smem[];
  h16*   Qb = (h16*)smem;
  h16*   Kb = (h16*)(smem + 57344);
  h16*   Vt = (h16*)(smem + 114688);
  h16*   Xl = (h16*)(smem + 180224);
  float* Sc = (float*)(smem + 180224);
  h16*   Pb = (h16*)(smem + 237568);
  h16*   AO = (h16*)(smem + 266240);
  float* red = (float*)(smem + 323584);

  int tid = threadIdx.x, lane = tid & 31, wave = tid >> 5;
  int lrow = lane & 15, hi8 = (lane & 16) ? 8 : 0;
  int b = blockIdx.x;
  float st = strengthP[0];

  // ---- stage 0 ----
  // Wave 0: DMA the 100x256 f16 X tile into LDS via the Tensor Data Mover.
  // Other waves: zero pad rows / Vt / Pb and prefetch the weight matrices.
  if (wave == 0) {
    unsigned long long ga =
        (unsigned long long)(uintptr_t)(X + (long long)b * 100 * 256);
    unsigned ldsoff = (unsigned)(uintptr_t)Xl;  // low 32 bits = LDS offset
    u32x4 g0;
    g0[0] = 1u;                                   // count=1, user D#
    g0[1] = ldsoff;                               // lds_addr
    g0[2] = (unsigned)(ga & 0xFFFFFFFFu);         // global_addr[31:0]
    g0[3] = (unsigned)((ga >> 32) & 0x1FFFFFFu) | (2u << 30);  // addr hi | type=2
    i32x8 g1;
    g1[0] = (int)(1u << 16);          // workgroup_mask=0, data_size=1 (2B)
    g1[1] = (int)(256u << 16);        // tensor_dim0[15:0]=256
    g1[2] = (int)(100u << 16);        // tensor_dim0 hi=0, tensor_dim1[15:0]=100
    g1[3] = (int)(256u << 16);        // tensor_dim1 hi=0, tile_dim0=256
    g1[4] = (int)100u;                // tile_dim1=100, tile_dim2=0
    g1[5] = (int)256u;                // tensor_dim0_stride[31:0]=256 elems
    g1[6] = 0;                        // stride hi, tensor_dim1_stride lo
    g1[7] = 0;
    i32x4 gz;
    gz[0] = 0; gz[1] = 0; gz[2] = 0; gz[3] = 0;
    i32x8 gz8;
    gz8[0] = 0; gz8[1] = 0; gz8[2] = 0; gz8[3] = 0;
    gz8[4] = 0; gz8[5] = 0; gz8[6] = 0; gz8[7] = 0;
    // clang-23 / amdgpu-toolchain 6-arg form:
    // (uint32x4 g0, int32x8 g1, int32x4, int32x4, int32x8, i32 cpol)
    __builtin_amdgcn_tensor_load_to_lds(g0, g1, gz, gz, gz8, 0);
    __builtin_amdgcn_s_wait_tensorcnt(0);
  } else {
    // zero the padding rows 100..111 of Xl (15 waves, 480 threads)
    int t = tid - 32;
    for (int i = t; i < 12 * 256; i += 480) Xl[112 * 256 - 12 * 256 + i] = (h16)0.f;
  }
  for (int i = tid; i < 256 * 128; i += 512) Vt[i] = (h16)0.f;
  for (int i = tid; i < 112 * 128; i += 512) Pb[i] = (h16)0.f;
  // warm L2 with the 7 weight matrices (each 256x256 f16 = 131072 B)
  __builtin_prefetch((const char*)Wq + tid * 256, 0, 3);
  __builtin_prefetch((const char*)Wk + tid * 256, 0, 3);
  __builtin_prefetch((const char*)Wv + tid * 256, 0, 3);
  __builtin_prefetch((const char*)Wo + tid * 256, 0, 3);
  __builtin_prefetch((const char*)Wl + tid * 256, 0, 3);
  __builtin_prefetch((const char*)Wg + tid * 256, 0, 3);
  __builtin_prefetch((const char*)Wz + tid * 256, 0, 3);
  __syncthreads();

  // ---- QKV projections: 3 * 7 * 16 = 336 tiles over 16 waves ----
  for (int j = wave; j < 336; j += 16) {
    int nt = j & 15, t2 = j >> 4, mt = t2 % 7, mat = t2 / 7;
    const h16* W = (mat == 0) ? Wq : ((mat == 1) ? Wk : Wv);
    const float* bi = (mat == 0) ? qb : ((mat == 1) ? kb : vb);
    int col = nt * 16 + lrow;
    float bv = bi[col];
    v8f acc;
#pragma unroll
    for (int r = 0; r < 8; r++) acc[r] = bv;
    int arow = mt * 16 + lrow;
    for (int kt = 0; kt < 8; kt++) {
      v16h a = loadA_rm(Xl, 256, arow, kt * 32 + hi8);
      v16h bb = loadB_prep(W, kt, nt, lane);
      acc = wmma16(a, bb, acc);
    }
    int rb0 = mt * 16 + hi8;
    if (mat == 2) {
#pragma unroll
      for (int r = 0; r < 8; r++) Vt[col * 128 + rb0 + r] = (h16)acc[r];
    } else {
      h16* dst = (mat == 0) ? Qb : Kb;
#pragma unroll
      for (int r = 0; r < 8; r++) dst[(rb0 + r) * 256 + col] = (h16)acc[r];
    }
  }
  __syncthreads();

  // ---- attention, head by head ----
  for (int h = 0; h < 8; ++h) {
    // scores = Q_h @ K_h^T / sqrt(32); 49 tiles
    for (int j = wave; j < 49; j += 16) {
      int mt = j / 7, nt = j % 7;
      v8f acc;
#pragma unroll
      for (int r = 0; r < 8; r++) acc[r] = 0.f;
      v16h a = loadA_rm(Qb + h * 32, 256, mt * 16 + lrow, hi8);
      v16h bb = loadA_rm(Kb + h * 32, 256, nt * 16 + lrow, hi8);
      acc = wmma16(a, bb, acc);
      int col = nt * 16 + lrow, rb0 = mt * 16 + hi8;
#pragma unroll
      for (int r = 0; r < 8; r++)
        Sc[(rb0 + r) * 112 + col] = acc[r] * 0.17677669529f;
    }
    __syncthreads();
    // masked softmax, wave per row
    for (int r = wave; r < 100; r += 16) {
      float mx = -1e30f;
      for (int c = lane; c < 100; c += 32) mx = fmaxf(mx, Sc[r * 112 + c]);
      mx = wmaxr(mx);
      float s = 0.f;
      for (int c = lane; c < 100; c += 32) s += expf(Sc[r * 112 + c] - mx);
      float inv = 1.f / wsum(s);
      for (int c = lane; c < 128; c += 32) {
        float p = (c < 100) ? expf(Sc[r * 112 + c] - mx) * inv : 0.f;
        Pb[r * 128 + c] = (h16)p;
      }
    }
    __syncthreads();
    // out_h = P @ V_h; 14 tiles (7 mt x 2 nt), K padded to 128
    if (wave < 14) {
      int mt = wave >> 1, nt = wave & 1;
      v8f acc;
#pragma unroll
      for (int r = 0; r < 8; r++) acc[r] = 0.f;
      for (int kt = 0; kt < 4; kt++) {
        v16h a = loadA_rm(Pb, 128, mt * 16 + lrow, kt * 32 + hi8);
        v16h bb = loadA_rm(Vt, 128, h * 32 + nt * 16 + lrow, kt * 32 + hi8);
        acc = wmma16(a, bb, acc);
      }
      int col = h * 32 + nt * 16 + lrow, rb0 = mt * 16 + hi8;
#pragma unroll
      for (int r = 0; r < 8; r++) AO[(rb0 + r) * 256 + col] = (h16)acc[r];
    }
    __syncthreads();
  }

  // ---- O projection: OUT = AO @ Wo + ob  (into Qb region) ----
  for (int j = wave; j < 112; j += 16) {
    int nt = j & 15, mt = j >> 4;
    int col = nt * 16 + lrow;
    float bv = ob[col];
    v8f acc;
#pragma unroll
    for (int r = 0; r < 8; r++) acc[r] = bv;
    int arow = mt * 16 + lrow;
    for (int kt = 0; kt < 8; kt++) {
      v16h a = loadA_rm(AO, 256, arow, kt * 32 + hi8);
      v16h bb = loadB_prep(Wo, kt, nt, lane);
      acc = wmma16(a, bb, acc);
    }
    int rb0 = mt * 16 + hi8;
#pragma unroll
    for (int r = 0; r < 8; r++) Qb[(rb0 + r) * 256 + col] = (h16)acc[r];
  }
  __syncthreads();

  // ---- QEL stage 1: x1 = out*((1-st) + st*sigmoid(roll(out) @ Wl)) ----
  for (int j = wave; j < 112; j += 16) {
    int nt = j & 15, mt = j >> 4;
    int col = nt * 16 + lrow;
    v8f acc;
#pragma unroll
    for (int r = 0; r < 8; r++) acc[r] = 0.f;
    int rowi = mt * 16 + lrow;
    int arow = (rowi == 0) ? 99 : (rowi - 1);  // roll(+1) on token axis
    for (int kt = 0; kt < 8; kt++) {
      v16h a = loadA_rm(Qb, 256, arow, kt * 32 + hi8);
      v16h bb = loadB_prep(Wl, kt, nt, lane);
      acc = wmma16(a, bb, acc);
    }
    int rb0 = mt * 16 + hi8;
#pragma unroll
    for (int r = 0; r < 8; r++) {
      int row = rb0 + r;
      float o = (float)Qb[row * 256 + col];
      float lg = 1.f / (1.f + expf(-acc[r]));
      Kb[row * 256 + col] = (h16)(o * (1.f - st) + o * st * lg);
    }
  }
  __syncthreads();

  // ---- QEL stage 2: x2 = x1*(1-st/2) + out*(st/2)*sigmoid(x1 @ Wg) ----
  h16* X2 = Vt;  // reuse Vt region as row-major 112x256
  for (int j = wave; j < 112; j += 16) {
    int nt = j & 15, mt = j >> 4;
    int col = nt * 16 + lrow;
    v8f acc;
#pragma unroll
    for (int r = 0; r < 8; r++) acc[r] = 0.f;
    int arow = mt * 16 + lrow;
    for (int kt = 0; kt < 8; kt++) {
      v16h a = loadA_rm(Kb, 256, arow, kt * 32 + hi8);
      v16h bb = loadB_prep(Wg, kt, nt, lane);
      acc = wmma16(a, bb, acc);
    }
    int rb0 = mt * 16 + hi8;
#pragma unroll
    for (int r = 0; r < 8; r++) {
      int row = rb0 + r;
      float o = (float)Qb[row * 256 + col];
      float x1 = (float)Kb[row * 256 + col];
      float gl = 1.f / (1.f + expf(-acc[r]));
      X2[row * 256 + col] = (h16)(x1 * (1.f - st * 0.5f) + o * st * 0.5f * gl);
    }
  }
  __syncthreads();

  // ---- final LayerNorm (FUSED into Kb) + entropy (deterministic) ----
  float entw = 0.f;
  for (int r = wave; r < 100; r += 16) {
    int c0 = lane * 8;
    float v[8], s = 0.f;
#pragma unroll
    for (int i = 0; i < 8; i++) { v[i] = (float)X2[r * 256 + c0 + i]; s += v[i]; }
    float mean = wsum(s) * (1.f / 256.f);
    float vs = 0.f;
#pragma unroll
    for (int i = 0; i < 8; i++) { float d = v[i] - mean; vs += d * d; }
    float rstd = rsqrtf(wsum(vs) * (1.f / 256.f) + 1e-5f);
    float f[8], amax = 0.f;
#pragma unroll
    for (int i = 0; i < 8; i++) {
      int c = c0 + i;
      f[i] = (v[i] - mean) * rstd * lng[c] + lnb[c];
      amax = fmaxf(amax, fabsf(f[i]));
      Kb[r * 256 + c] = (h16)f[i];
    }
    amax = wmaxr(amax);
    float inv = 1.f / (amax + 1e-8f);
    float ai[8], es = 0.f;
#pragma unroll
    for (int i = 0; i < 8; i++) { ai[i] = fabsf(f[i]) * inv; es += expf(ai[i]); }
    float invS = 1.f / wsum(es);
    float el = 0.f;
#pragma unroll
    for (int i = 0; i < 8; i++) {
      float p = expf(ai[i]) * invS;
      el += p * logf(p + 1e-8f);
    }
    entw += -wsum(el);
  }
  if (lane == 0) red[wave] = entw;
  __syncthreads();
  if (tid == 0) {
    float t = 0.f;
    for (int w = 0; w < 16; w++) t += red[w];
    entPart[b] = t;
  }
  __syncthreads();

  // ---- output projection: d_out = FUSED @ Wz + zb ----
  for (int j = wave; j < 112; j += 16) {
    int nt = j & 15, mt = j >> 4;
    int col = nt * 16 + lrow;
    float bv = zb[col];
    v8f acc;
#pragma unroll
    for (int r = 0; r < 8; r++) acc[r] = bv;
    int arow = mt * 16 + lrow;
    for (int kt = 0; kt < 8; kt++) {
      v16h a = loadA_rm(Kb, 256, arow, kt * 32 + hi8);
      v16h bb = loadB_prep(Wz, kt, nt, lane);
      acc = wmma16(a, bb, acc);
    }
    int rb0 = mt * 16 + hi8;
#pragma unroll
    for (int r = 0; r < 8; r++) {
      int row = rb0 + r;
      if (row < 100) out[((long long)b * 100 + row) * 256 + col] = acc[r];
    }
  }
}

// ------------------------------------------------- entropy finalization -----
__global__ void ent_final(const float* __restrict__ part, float* __restrict__ o,
                          int n, float scale) {
  __shared__ float sh[256];
  float s = 0.f;
  for (int i = threadIdx.x; i < n; i += 256) s += part[i];
  sh[threadIdx.x] = s;
  __syncthreads();
  for (int off = 128; off > 0; off >>= 1) {
    if (threadIdx.x < off) sh[threadIdx.x] += sh[threadIdx.x + off];
    __syncthreads();
  }
  if (threadIdx.x == 0) o[0] = sh[0] * scale;
}

// ------------------------------------------------------------------- host ---

extern "C" void kernel_launch(void* const* d_in, const int* in_sizes, int n_in,
                              void* d_out, int out_size, void* d_ws,
                              size_t ws_size, hipStream_t stream) {
  const float* vis = (const float*)d_in[0];
  const float* sen = (const float*)d_in[1];
  const float* spe = (const float*)d_in[2];
  const float* visW = (const float*)d_in[3];
  const float* visb = (const float*)d_in[4];
  const float* visg = (const float*)d_in[5];
  const float* visbeta = (const float*)d_in[6];
  const float* senW = (const float*)d_in[7];
  const float* senb = (const float*)d_in[8];
  const float* seng = (const float*)d_in[9];
  const float* senbeta = (const float*)d_in[10];
  const float* speW = (const float*)d_in[11];
  const float* speb = (const float*)d_in[12];
  const float* speg = (const float*)d_in[13];
  const float* spebeta = (const float*)d_in[14];
  const float* qW = (const float*)d_in[15];
  const float* qb = (const float*)d_in[16];
  const float* kW = (const float*)d_in[17];
  const float* kb = (const float*)d_in[18];
  const float* vW = (const float*)d_in[19];
  const float* vb = (const float*)d_in[20];
  const float* oW = (const float*)d_in[21];
  const float* ob = (const float*)d_in[22];
  const float* lW = (const float*)d_in[23];
  const float* gW = (const float*)d_in[24];
  const float* strength = (const float*)d_in[25];
  const float* lng = (const float*)d_in[26];
  const float* lnb = (const float*)d_in[27];
  const float* zW = (const float*)d_in[28];
  const float* zb = (const float*)d_in[29];
  const float* pos = (const float*)d_in[30];
  float* outp = (float*)d_out;

  char* ws = (char*)d_ws;
  h16* WvisP = (h16*)(ws + 0);        // 512x256
  h16* WsenP = (h16*)(ws + 262144);   // 64x256
  h16* WspeP = (h16*)(ws + 294912);   // 128x256
  h16* WqP = (h16*)(ws + 360448);
  h16* WkP = (h16*)(ws + 491520);
  h16* WvP = (h16*)(ws + 622592);
  h16* WoP = (h16*)(ws + 753664);
  h16* WlP = (h16*)(ws + 884736);
  h16* WgP = (h16*)(ws + 1015808);
  h16* WzP = (h16*)(ws + 1146880);
  h16* X = (h16*)(ws + 1277952);              // 2048*100*256 f16
  float* entPart = (float*)(ws + 106135552);  // 2048 f32

  // --- weight repack (f16, WMMA B-fragment layout) ---
  prep_w<<<(512 * 256 + 255) / 256, 256, 0, stream>>>(visW, WvisP, 512);
  prep_w<<<(64 * 256 + 255) / 256, 256, 0, stream>>>(senW, WsenP, 64);
  prep_w<<<(128 * 256 + 255) / 256, 256, 0, stream>>>(speW, WspeP, 128);
  prep_w<<<(256 * 256 + 255) / 256, 256, 0, stream>>>(qW, WqP, 256);
  prep_w<<<(256 * 256 + 255) / 256, 256, 0, stream>>>(kW, WkP, 256);
  prep_w<<<(256 * 256 + 255) / 256, 256, 0, stream>>>(vW, WvP, 256);
  prep_w<<<(256 * 256 + 255) / 256, 256, 0, stream>>>(oW, WoP, 256);
  prep_w<<<(256 * 256 + 255) / 256, 256, 0, stream>>>(lW, WlP, 256);
  prep_w<<<(256 * 256 + 255) / 256, 256, 0, stream>>>(gW, WgP, 256);
  prep_w<<<(256 * 256 + 255) / 256, 256, 0, stream>>>(zW, WzP, 256);

  // --- modal encoders -> X (f16) ---
  encoder_kernel<<<2048 * 50 / 16, 512, 0, stream>>>(vis, 512, 0, 50, WvisP,
                                                     visb, visg, visbeta, pos, X);
  encoder_kernel<<<2048 * 25 / 16, 512, 0, stream>>>(sen, 64, 50, 25, WsenP,
                                                     senb, seng, senbeta, pos, X);
  encoder_kernel<<<2048 * 25 / 16, 512, 0, stream>>>(spe, 128, 75, 25, WspeP,
                                                     speb, speg, spebeta, pos, X);

  // --- fused attention + QEL + LN + entropy + out projection ---
  fused_kernel<<<2048, 512, FUSED_SMEM, stream>>>(
      X, WqP, WkP, WvP, WoP, WlP, WgP, WzP, qb, kb, vb, ob, strength, lng, lnb,
      zb, outp, entPart);

  // --- entropy mean over B*N = 204800 rows ---
  ent_final<<<1, 256, 0, stream>>>(entPart, outp + (out_size - 1), 2048,
                                   1.0f / 204800.0f);
}
